// Attention_58351425683484
// MI455X (gfx1250) — matrix operand order repaired and checked by reading, not compiled, not run on hardware
//
#include <hip/hip_runtime.h>
#include <hip/hip_bf16.h>

// B=4, S=2048, D=1024, N=16 heads, H=64.
// Matmuls: v_wmma_f32_16x16x32_f16. Tile staging: TDM tensor_load_to_lds (+ LDS pad),
// double-buffered in the GEMMs, tracked with s_wait_tensorcnt. Fragment loads batched
// so WMMAs issue in runs behind a single s_wait_dscnt.

typedef __attribute__((ext_vector_type(16))) _Float16 v16h;
typedef __attribute__((ext_vector_type(8)))  float    v8f;
typedef __attribute__((ext_vector_type(4)))  unsigned int uvec4;
typedef __attribute__((ext_vector_type(8)))  int          ivec8;
typedef __attribute__((ext_vector_type(4)))  int          ivec4;

#define WMMA_F16(a, b, c) \
  __builtin_amdgcn_wmma_f32_16x16x32_f16(false, (a), false, (b), (short)0, (c), false, false)

#if __has_include(<hip/amd_detail/amd_gfx1250_TDM.h>)
#define TDM_6ARGS 1
#endif

// 16-bit A-matrix 16x32 fragment K-offset for VGPR v (contiguous, matches B):
// lanes 0-15 : v -> K=2v,2v+1 (0..15)
// lanes 16-31: +16            (16..31)
__device__ __forceinline__ int a_koff(int hi, int v) {
  return 2 * v + (hi ? 16 : 0);
}
// 16-bit B-matrix 32x16 fragment: lanes 0-15 hold K=0..15 (v -> 2v,2v+1), lanes 16-31 K=16..31.

__device__ __forceinline__ v8f zero8() {
  v8f z = {0.f, 0.f, 0.f, 0.f, 0.f, 0.f, 0.f, 0.f};
  return z;
}

// ---------------------------------------------------------------- TDM 2D tile load
// Loads a tile_d0 x tile_d1 tile of 2-byte elements (row stride d0_stride elements in
// memory) into LDS at lds_off, inserting pad_amount dwords after every pad_interval
// dwords (encodings per D# group1: interval enc k -> 2^(k+1) dwords; amount enc a -> a+1).
__device__ __forceinline__ void tdm_load_2d(unsigned lds_off, const void* gaddr,
                                            unsigned tile_d0, unsigned tile_d1,
                                            unsigned d0_stride,
                                            unsigned pad_interval_enc,
                                            unsigned pad_amount_enc) {
  unsigned long long ga = (unsigned long long)(size_t)gaddr;
  uvec4 g0;
  g0[0] = 1u;                                        // count=1 (valid), user mode
  g0[1] = lds_off;                                   // LDS byte address
  g0[2] = (unsigned)(ga & 0xffffffffu);              // global_addr[31:0]
  g0[3] = (unsigned)((ga >> 32) & 0x1ffffffu) | (2u << 30);  // addr[56:32], type=2
  ivec8 g1;
  g1[0] = (int)((1u << 16) |                         // data_size = 2 bytes
                (1u << 20) |                         // pad_enable
                (pad_interval_enc << 22) | (pad_amount_enc << 25));
  g1[1] = (int)((tile_d0 & 0xffffu) << 16);          // tensor_dim0[15:0] (== tile_d0)
  g1[2] = (int)(((tile_d0 >> 16) & 0xffffu) | ((tile_d1 & 0xffffu) << 16)); // t_dim1 lo
  g1[3] = (int)(((tile_d1 >> 16) & 0xffffu) | ((tile_d0 & 0xffffu) << 16)); // tile_dim0
  g1[4] = (int)(tile_d1 & 0xffffu);                  // tile_dim1 ; tile_dim2 = 0
  g1[5] = (int)d0_stride;                            // tensor_dim0_stride[31:0]
  g1[6] = 0;                                         // stride hi / dim1_stride lo
  g1[7] = 0;
  ivec4 gz = {0, 0, 0, 0};
#ifdef TDM_6ARGS
  ivec8 gz8 = {0, 0, 0, 0, 0, 0, 0, 0};
  __builtin_amdgcn_tensor_load_to_lds(g0, g1, gz, gz, gz8, 0);
#else
  __builtin_amdgcn_tensor_load_to_lds(g0, g1, gz, gz, 0);
#endif
}

__device__ __forceinline__ unsigned lds_addr_of(const void* p) {
  return (unsigned)(unsigned long long)(size_t)p;  // flat LDS aperture: addr[31:0] = LDS offset
}

// ---------------------------------------------------------------- converts
__global__ void cvt_f32_to_f16(const float* __restrict__ in, _Float16* __restrict__ out, int n) {
  for (int i = blockIdx.x * blockDim.x + threadIdx.x; i < n; i += gridDim.x * blockDim.x)
    out[i] = (_Float16)in[i];
}

// out[c*rows + r] = in[r*cols + c]  (weights -> [N][K] f16 so K is contiguous)
__global__ void transpose_f32_to_f16(const float* __restrict__ in, _Float16* __restrict__ out,
                                     int rows, int cols) {
  int n = rows * cols;
  for (int i = blockIdx.x * blockDim.x + threadIdx.x; i < n; i += gridDim.x * blockDim.x) {
    int r = i / cols, c = i - r * cols;
    out[(size_t)c * rows + r] = (_Float16)in[i];
  }
}

// ---------------------------------------------------------------- shared GEMM core
// One 128x128 tile of A[*][1024] * BT[*][1024]^T with TDM double-buffering.
// Wave `wave` owns rows [16*wave, 16*wave+16); acc[j] covers cols [16j,16j+16).
__device__ __forceinline__ void gemm_core_128x128(const _Float16* A, const _Float16* BT,
                                                  int mbase, int nbase, int wave, int lane,
                                                  _Float16 (*As)[128 * 40],
                                                  _Float16 (*Bs)[128 * 40], v8f acc[8]) {
  const int ln = lane & 15, hi = lane >> 4;
#pragma unroll
  for (int j = 0; j < 8; ++j) acc[j] = zero8();

  // prologue: stage k0 = 0 tiles into buffer 0
  if (wave == 0) {
    tdm_load_2d(lds_addr_of(&As[0][0]), A + (size_t)mbase * 1024, 32, 128, 1024, 3, 3);
    tdm_load_2d(lds_addr_of(&Bs[0][0]), BT + (size_t)nbase * 1024, 32, 128, 1024, 3, 3);
  }

  for (int k0 = 0; k0 < 1024; k0 += 32) {
    const int buf = (k0 >> 5) & 1;
    if (wave == 0) {
      if (k0 + 32 < 1024) {  // prefetch next tile into the other buffer
        tdm_load_2d(lds_addr_of(&As[buf ^ 1][0]),
                    A + (size_t)mbase * 1024 + k0 + 32, 32, 128, 1024, 3, 3);
        tdm_load_2d(lds_addr_of(&Bs[buf ^ 1][0]),
                    BT + (size_t)nbase * 1024 + k0 + 32, 32, 128, 1024, 3, 3);
        __builtin_amdgcn_s_wait_tensorcnt(2);  // oldest pair (current tile) complete
      } else {
        __builtin_amdgcn_s_wait_tensorcnt(0);
      }
    }
    __syncthreads();

    v16h af;
    unsigned* ap = (unsigned*)&af;
    const int arow = wave * 16 + ln;
#pragma unroll
    for (int v = 0; v < 8; ++v)
      ap[v] = *(const unsigned*)&As[buf][arow * 40 + a_koff(hi, v)];

    // batch 4 B-fragments, then 4 back-to-back WMMAs (one dscnt wait per group)
#pragma unroll
    for (int jg = 0; jg < 2; ++jg) {
      v16h bfr[4];
#pragma unroll
      for (int u = 0; u < 4; ++u) {
        unsigned* bp = (unsigned*)&bfr[u];
        const int brow = (jg * 4 + u) * 16 + ln;
#pragma unroll
        for (int v = 0; v < 8; ++v)
          bp[v] = *(const unsigned*)&Bs[buf][brow * 40 + hi * 16 + 2 * v];
      }
#pragma unroll
      for (int u = 0; u < 4; ++u)
        acc[jg * 4 + u] = WMMA_F16(af, bfr[u], acc[jg * 4 + u]);
    }
    __syncthreads();  // all waves done with this buffer before TDM may overwrite it
  }
}

// ---------------------------------------------------------------- QKV GEMM
__global__ __launch_bounds__(256) void gemm_qkv(const _Float16* __restrict__ A,
                                                const _Float16* __restrict__ BT,
                                                const float* __restrict__ bias,
                                                _Float16* __restrict__ qo,
                                                _Float16* __restrict__ ko,
                                                _Float16* __restrict__ vo) {
  __shared__ _Float16 As[2][128 * 40];
  __shared__ _Float16 Bs[2][128 * 40];
  const int tid = threadIdx.x;
  const int wave = tid >> 5, lane = tid & 31;
  const int ln = lane & 15, hi = lane >> 4;
  const int mbase = blockIdx.x * 128;
  const int nbase = blockIdx.y * 128;

  v8f acc[8];
  gemm_core_128x128(A, BT, mbase, nbase, wave, lane, As, Bs, acc);

  const int which = nbase >> 10;  // each 128-col tile lies entirely in one of q/k/v
  _Float16* dst = (which == 0) ? qo : (which == 1) ? ko : vo;
#pragma unroll
  for (int j = 0; j < 8; ++j) {
    const float* cp = (const float*)&acc[j];
#pragma unroll
    for (int r = 0; r < 8; ++r) {
      int m = mbase + wave * 16 + r + 8 * hi;  // C layout: M = reg + 8*(lane>=16)
      int c = nbase + j * 16 + ln;             // N = lane%16
      float val = cp[r] + bias[c];
      int dd = c & 1023;
      int nh = dd >> 6, hh = dd & 63;
      int bb = m >> 11, ss = m & 2047;
      dst[(((size_t)(bb * 16 + nh)) * 2048 + ss) * 64 + hh] = (_Float16)val;
    }
  }
}

// ---------------------------------------------------------------- output projection
__global__ __launch_bounds__(256) void gemm_proj(const _Float16* __restrict__ A,
                                                 const _Float16* __restrict__ BT,
                                                 const float* __restrict__ bias,
                                                 float* __restrict__ out) {
  __shared__ _Float16 As[2][128 * 40];
  __shared__ _Float16 Bs[2][128 * 40];
  const int tid = threadIdx.x;
  const int wave = tid >> 5, lane = tid & 31;
  const int ln = lane & 15, hi = lane >> 4;
  const int mbase = blockIdx.x * 128;
  const int nbase = blockIdx.y * 128;

  v8f acc[8];
  gemm_core_128x128(A, BT, mbase, nbase, wave, lane, As, Bs, acc);

#pragma unroll
  for (int j = 0; j < 8; ++j) {
    const float* cp = (const float*)&acc[j];
#pragma unroll
    for (int r = 0; r < 8; ++r) {
      int m = mbase + wave * 16 + r + 8 * hi;
      int c = nbase + j * 16 + ln;
      out[(size_t)m * 1024 + c] = cp[r] + bias[c];
    }
  }
}

// ---------------------------------------------------------------- flash attention
// grid = B*N*(S/128) = 1024 blocks, 256 threads (8 waves); wave w owns Q rows [16w,16w+16)
__global__ __launch_bounds__(256) void attn_kernel(const _Float16* __restrict__ qh,
                                                   const _Float16* __restrict__ kh,
                                                   const _Float16* __restrict__ vh,
                                                   const float* __restrict__ amask,
                                                   _Float16* __restrict__ out) {
  extern __shared__ char smraw[];
  _Float16* Kt = (_Float16*)smraw;   // [128][72]  key-major, dims contiguous (TDM + pad)
  _Float16* Vt = Kt + 128 * 72;      // [64][136]  headdim-major, keys contiguous
  _Float16* Pb = Vt + 64 * 136;      // 8 waves x [16][136] probs (wave-private)

  const int tid = threadIdx.x;
  const int wave = tid >> 5, lane = tid & 31;
  const int ln = lane & 15, hi = lane >> 4;
  const int blk = blockIdx.x;
  const int qtile = blk & 15;
  const int bn = blk >> 4;
  const int n = bn & 15, bq = bn >> 4;
  const size_t headoff = ((size_t)(bq * 16 + n)) * 2048 * 64;
  const _Float16* Q = qh + headoff;
  const _Float16* K = kh + headoff;
  const _Float16* V = vh + headoff;
  const int q0 = qtile * 128;
  _Float16* P = Pb + wave * (16 * 136);
  const unsigned kt_lds = lds_addr_of(Kt);

  // Q strip (16x64) kept in registers as two 16x32 A fragments
  v16h qa[2];
  {
    const int qrow = q0 + wave * 16 + ln;
#pragma unroll
    for (int c = 0; c < 2; ++c) {
      unsigned* qp = (unsigned*)&qa[c];
#pragma unroll
      for (int v = 0; v < 8; ++v)
        qp[v] = *(const unsigned*)(Q + (size_t)qrow * 64 + 32 * c + a_koff(hi, v));
    }
  }

  v8f acc[4];
#pragma unroll
  for (int j = 0; j < 4; ++j) acc[j] = zero8();
  float mrow[8], lrow[8];
#pragma unroll
  for (int r = 0; r < 8; ++r) { mrow[r] = -3.0e38f; lrow[r] = 0.f; }

  for (int t = 0; t < 16; ++t) {
    // K tile 128x64 -> Kt[128][72] via TDM (pad 4 dwords per 32-dword row)
    if (wave == 0)
      tdm_load_2d(kt_lds, K + (size_t)t * 128 * 64, 64, 128, 64, 4, 3);
    // V tile 128x64 transposed -> Vt[64][136] (manual: TDM cannot transpose)
#pragma unroll
    for (int it = 0; it < 16; ++it) {
      int chunk = tid + it * 256;           // 4096 dword chunks
      int row = chunk >> 5;                 // key pos
      int col = (chunk & 31) * 2;           // head dim pair
      unsigned d = *(const unsigned*)(V + (size_t)(t * 128 + row) * 64 + col);
      unsigned short* vt16 = (unsigned short*)Vt;
      vt16[(col + 0) * 136 + row] = (unsigned short)(d & 0xffffu);
      vt16[(col + 1) * 136 + row] = (unsigned short)(d >> 16);
    }
    if (wave == 0) __builtin_amdgcn_s_wait_tensorcnt(0);
    __syncthreads();

    // scores S = Q*K^T / 8 - (1-mask)*1e-6 ; 16x128 per wave.
    // Key sub-tiles processed in pairs: 4 batched B-frag loads, then 4 WMMAs.
    v8f sc[8];
#pragma unroll
    for (int jg = 0; jg < 4; ++jg) {
      const int j0 = jg * 2, j1 = jg * 2 + 1;
      const int key0 = j0 * 16 + ln, key1 = j1 * 16 + ln;
      v16h bA0, bA1, bB0, bB1;
      unsigned* a0 = (unsigned*)&bA0;
      unsigned* a1 = (unsigned*)&bA1;
      unsigned* b0 = (unsigned*)&bB0;
      unsigned* b1 = (unsigned*)&bB1;
#pragma unroll
      for (int v = 0; v < 8; ++v) {
        a0[v] = *(const unsigned*)&Kt[key0 * 72 + hi * 16 + 2 * v];
        a1[v] = *(const unsigned*)&Kt[key0 * 72 + 32 + hi * 16 + 2 * v];
        b0[v] = *(const unsigned*)&Kt[key1 * 72 + hi * 16 + 2 * v];
        b1[v] = *(const unsigned*)&Kt[key1 * 72 + 32 + hi * 16 + 2 * v];
      }
      v8f z0 = zero8(), z1 = zero8();
      z0 = WMMA_F16(qa[0], bA0, z0);
      z1 = WMMA_F16(qa[0], bB0, z1);
      z0 = WMMA_F16(qa[1], bA1, z0);
      z1 = WMMA_F16(qa[1], bB1, z1);
      const float pen0 = (1.0f - amask[(size_t)bq * 2048 + t * 128 + key0]) * 1e-6f;
      const float pen1 = (1.0f - amask[(size_t)bq * 2048 + t * 128 + key1]) * 1e-6f;
      float* zp0 = (float*)&z0;
      float* zp1 = (float*)&z1;
#pragma unroll
      for (int r = 0; r < 8; ++r) {
        zp0[r] = zp0[r] * 0.125f - pen0;
        zp1[r] = zp1[r] * 0.125f - pen1;
      }
      sc[j0] = z0;
      sc[j1] = z1;
    }

    // online softmax: rows live at (reg r + 8*hi), cols across 16 lanes of the half
    float mx[8];
#pragma unroll
    for (int r = 0; r < 8; ++r) mx[r] = ((const float*)&sc[0])[r];
#pragma unroll
    for (int j = 1; j < 8; ++j) {
      const float* zp = (const float*)&sc[j];
#pragma unroll
      for (int r = 0; r < 8; ++r) mx[r] = fmaxf(mx[r], zp[r]);
    }
#pragma unroll
    for (int off = 1; off <= 8; off <<= 1) {  // stays inside each 16-lane half
#pragma unroll
      for (int r = 0; r < 8; ++r) mx[r] = fmaxf(mx[r], __shfl_xor(mx[r], off, 32));
    }
    float al[8], rs[8];
#pragma unroll
    for (int r = 0; r < 8; ++r) {
      float mnew = fmaxf(mrow[r], mx[r]);
      al[r] = __expf(mrow[r] - mnew);
      mrow[r] = mnew;
      rs[r] = 0.f;
    }
#pragma unroll
    for (int j = 0; j < 8; ++j) {
      const float* zp = (const float*)&sc[j];
#pragma unroll
      for (int r = 0; r < 8; ++r) {
        float p = __expf(zp[r] - mrow[r]);
        rs[r] += p;
        P[(r + 8 * hi) * 136 + j * 16 + ln] = (_Float16)p;  // C layout -> A layout via LDS
      }
    }
#pragma unroll
    for (int off = 1; off <= 8; off <<= 1) {
#pragma unroll
      for (int r = 0; r < 8; ++r) rs[r] += __shfl_xor(rs[r], off, 32);
    }
#pragma unroll
    for (int r = 0; r < 8; ++r) lrow[r] = lrow[r] * al[r] + rs[r];
#pragma unroll
    for (int j2 = 0; j2 < 4; ++j2) {
      float* apx = (float*)&acc[j2];
#pragma unroll
      for (int r = 0; r < 8; ++r) apx[r] *= al[r];
    }
    __syncthreads();  // P writes visible before P@V

    // O += P (16x128) @ V (128x64): per k-chunk load pa + all 4 vb frags, then 4 WMMAs
#pragma unroll
    for (int c2 = 0; c2 < 4; ++c2) {
      v16h pa;
      unsigned* pp = (unsigned*)&pa;
#pragma unroll
      for (int v = 0; v < 8; ++v)
        pp[v] = *(const unsigned*)&P[ln * 136 + 32 * c2 + a_koff(hi, v)];
      v16h vb[4];
#pragma unroll
      for (int j2 = 0; j2 < 4; ++j2) {
        unsigned* vp = (unsigned*)&vb[j2];
        const int hd = j2 * 16 + ln;
#pragma unroll
        for (int v = 0; v < 8; ++v)
          vp[v] = *(const unsigned*)&Vt[hd * 136 + 32 * c2 + hi * 16 + 2 * v];
      }
#pragma unroll
      for (int j2 = 0; j2 < 4; ++j2)
        acc[j2] = WMMA_F16(pa, vb[j2], acc[j2]);
    }
    __syncthreads();  // done reading Kt/Vt before next tile overwrites
  }

  // normalize and store f16 attn_out in [B,S,N,H] (== [B,S,D] flat)
#pragma unroll
  for (int j2 = 0; j2 < 4; ++j2) {
    const float* apx = (const float*)&acc[j2];
#pragma unroll
    for (int r = 0; r < 8; ++r) {
      const int m = q0 + wave * 16 + r + 8 * hi;
      const float o = apx[r] / lrow[r];
      out[((size_t)(bq * 2048 + m)) * 1024 + n * 64 + j2 * 16 + ln] = (_Float16)o;
    }
  }
}

// ---------------------------------------------------------------- launcher
extern "C" void kernel_launch(void* const* d_in, const int* in_sizes, int n_in,
                              void* d_out, int out_size, void* d_ws, size_t ws_size,
                              hipStream_t stream) {
  const float* x      = (const float*)d_in[0];
  const float* amask  = (const float*)d_in[1];
  const float* w_attn = (const float*)d_in[2];
  const float* b_attn = (const float*)d_in[3];
  const float* w_proj = (const float*)d_in[4];
  const float* b_proj = (const float*)d_in[5];
  float* out = (float*)d_out;

  // workspace layout (bytes): total 92,274,688
  char* ws = (char*)d_ws;
  _Float16* xh  = (_Float16*)(ws + 0);          // 8192*1024 f16 = 16 MiB
  _Float16* wAT = (_Float16*)(ws + 16777216);   // 3072*1024 f16 =  6 MiB
  _Float16* wPT = (_Float16*)(ws + 23068672);   // 1024*1024 f16 =  2 MiB
  _Float16* qh  = (_Float16*)(ws + 25165824);   // [B,N,S,H] f16 = 16 MiB
  _Float16* kh  = (_Float16*)(ws + 41943040);
  _Float16* vh  = (_Float16*)(ws + 58720256);
  _Float16* ah  = (_Float16*)(ws + 75497472);   // attn out [B,S,D] f16

  cvt_f32_to_f16<<<4096, 256, 0, stream>>>(x, xh, 8192 * 1024);
  transpose_f32_to_f16<<<4096, 256, 0, stream>>>(w_attn, wAT, 1024, 3072);
  transpose_f32_to_f16<<<2048, 256, 0, stream>>>(w_proj, wPT, 1024, 1024);

  gemm_qkv<<<dim3(64, 24), 256, 0, stream>>>(xh, wAT, b_attn, qh, kh, vh);

  // dynamic LDS: 128*72 + 64*136 + 8*16*136 halves = 70,656 bytes
  attn_kernel<<<1024, 256, 70656, stream>>>(qh, kh, vh, amask, ah);

  gemm_proj<<<dim3(64, 8), 256, 0, stream>>>(ah, wPT, b_proj, out);
}
